// PPGCN_36223754174750
// MI455X (gfx1250) — matrix operand order repaired and checked by reading, not compiled, not run on hardware
//
#include <hip/hip_runtime.h>
#include <math.h>

// ---------------------------------------------------------------------------
// Problem constants (from the reference)
// ---------------------------------------------------------------------------
#define RREL 3
#define NNODES 8192
#define BB 4096
#define IN_DIM 256
#define HID 128
#define OUTD 64
#define EPS 1e-5f

typedef __attribute__((ext_vector_type(16))) __bf16 v16bf;
typedef __attribute__((ext_vector_type(8)))  __bf16 v8bf;
typedef __attribute__((ext_vector_type(8)))  float  v8f;

__device__ __forceinline__ float eluf(float x) {
    return x > 0.f ? x : expm1f(x);
}

// ---------------------------------------------------------------------------
// Column mean / rstd over `rows` rows (optionally gathered via idx).
// One block per column.
// ---------------------------------------------------------------------------
__global__ void __launch_bounds__(256)
k_col_stats(const float* __restrict__ src, const int* __restrict__ idx,
            int rows, int cols, float* __restrict__ mean, float* __restrict__ rstd)
{
    int c = blockIdx.x;
    int tid = threadIdx.x;
    __shared__ float ssum[256], ssq[256];
    float s = 0.f, q = 0.f;
    for (int i = tid; i < rows; i += 256) {
        int rr = idx ? idx[i] : i;
        float x = src[(size_t)rr * cols + c];
        s += x; q += x * x;
    }
    ssum[tid] = s; ssq[tid] = q;
    __syncthreads();
    for (int st = 128; st > 0; st >>= 1) {
        if (tid < st) { ssum[tid] += ssum[tid + st]; ssq[tid] += ssq[tid + st]; }
        __syncthreads();
    }
    if (tid == 0) {
        float m = ssum[0] / (float)rows;
        float v = ssq[0] / (float)rows - m * m;
        mean[c] = m;
        rstd[c] = rsqrtf(v + EPS);
    }
}

// xn[i,c] = bf16( (features[bn[i],c]-mean)*rstd*g + b ),  one block per row
__global__ void __launch_bounds__(256)
k_bn_gather_bf16(const float* __restrict__ feat, const int* __restrict__ bn,
                 const float* __restrict__ mean, const float* __restrict__ rstd,
                 const float* __restrict__ g, const float* __restrict__ b,
                 __bf16* __restrict__ dst)
{
    int i = blockIdx.x;
    int c = threadIdx.x;                 // IN_DIM == 256 == blockDim.x
    int src = bn[i];
    float x = feat[(size_t)src * IN_DIM + c];
    float y = (x - mean[c]) * rstd[c] * g[c] + b[c];
    dst[(size_t)i * IN_DIM + c] = (__bf16)y;
}

// wt[n,k] = bf16(w[k,n]) : store weights column-major (K-contiguous) in bf16
__global__ void __launch_bounds__(256)
k_transpose_bf16(const float* __restrict__ w, __bf16* __restrict__ wt, int K, int Nn)
{
    int n = blockIdx.x;
    for (int k = threadIdx.x; k < K; k += 256)
        wt[(size_t)n * K + k] = (__bf16)w[(size_t)k * Nn + n];
}

// ---------------------------------------------------------------------------
// Gather batch adjacency for relation r into bf16 (values 0/1/2: exact in
// bf16), add identity, compute dinv[i] = rsqrt(rowsum). bn cached in LDS.
// One block per output row.
// ---------------------------------------------------------------------------
__global__ void __launch_bounds__(256)
k_gather_adj(const float* __restrict__ adj, int r, const int* __restrict__ bn,
             __bf16* __restrict__ bA, float* __restrict__ dinv)
{
    __shared__ int sbn[BB];
    __shared__ float ssum[256];
    int tid = threadIdx.x;
    for (int j = tid; j < BB; j += 256) sbn[j] = bn[j];
    __syncthreads();

    int i = blockIdx.x;
    const float* arow = adj + (size_t)r * NNODES * NNODES + (size_t)sbn[i] * NNODES;
    float s = 0.f;
    for (int j = tid; j < BB; j += 256) {
        float v = arow[sbn[j]];
        if (j == i) v += 1.f;            // + identity
        s += v;
        bA[(size_t)i * BB + j] = (__bf16)v;
    }
    ssum[tid] = s;
    __syncthreads();
    for (int st = 128; st > 0; st >>= 1) {
        if (tid < st) ssum[tid] += ssum[tid + st];
        __syncthreads();
    }
    if (tid == 0) dinv[i] = rsqrtf(ssum[0]);
}

// Zst[n, j] = bf16( dinv[j] * Xw1[j, n] )   (transposed + row-prescaled RHS)
__global__ void __launch_bounds__(128)
k_build_zst(const float* __restrict__ Xw1, const float* __restrict__ dinv,
            __bf16* __restrict__ Zst)
{
    int j = blockIdx.x;
    int n = threadIdx.x;                 // HID == 128
    float d = dinv[j];
    Zst[(size_t)n * BB + j] = (__bf16)(d * Xw1[(size_t)j * HID + n]);
}

// ---------------------------------------------------------------------------
// WMMA fragment helpers (bf16 16x16x32, A/B lanes hold contiguous k = 16*(lane/16) + s)
// ---------------------------------------------------------------------------
union UA { v16bf v; v8bf h[2]; };

// ---------------------------------------------------------------------------
// Single-tile bf16 WMMA GEMM:  acc(f32) = A[MxK](row-major) * Bt[NxK]^T
// Block: 256 threads = 8 waves; wave w -> row tile (blockIdx.y*8+w),
// col tile = blockIdx.x. Requires M%128==0, N%16==0, K%32==0.
// EPI 0: C[row*ldc+col] = acc                               (float out)
// EPI 2: T[col*ldc+row] = bf16(dinv[row]*acc)               (bf16 out, transposed)
// ---------------------------------------------------------------------------
template <int EPI>
__global__ void __launch_bounds__(256)
k_gemm_wmma(const __bf16* __restrict__ A, const __bf16* __restrict__ Bt,
            int K, const float* __restrict__ dinv, void* __restrict__ Cout, int ldc)
{
    int tid  = threadIdx.x;
    int wave = tid >> 5;
    int lane = tid & 31;
    int hlf  = (lane >> 4) & 1;
    int r16  = lane & 15;

    int tileRow = (blockIdx.y * 8 + wave) * 16;
    int tileCol = blockIdx.x * 16;

    v8f acc = {};
    const __bf16* pa0 = A  + (size_t)(tileRow + r16) * K + hlf * 16;
    const __bf16* pb0 = Bt + (size_t)(tileCol + r16) * K + hlf * 16;

#pragma unroll 2
    for (int k0 = 0; k0 < K; k0 += 32) {
        UA ua, ub;
        ua.h[0] = *(const v8bf*)(pa0 + k0);
        ua.h[1] = *(const v8bf*)(pa0 + k0 + 8);
        ub.h[0] = *(const v8bf*)(pb0 + k0);
        ub.h[1] = *(const v8bf*)(pb0 + k0 + 8);
        acc = __builtin_amdgcn_wmma_f32_16x16x32_bf16(
            false, ua.v, false, ub.v, (short)0, acc, false, false);
    }

    int col = tileCol + r16;
#pragma unroll
    for (int v = 0; v < 8; ++v) {
        int row = tileRow + v + hlf * 8;
        float a = acc[v];
        if (EPI == 0) {
            ((float*)Cout)[(size_t)row * ldc + col] = a;
        } else { // EPI == 2: transposed + dinv-row-scaled bf16 store
            float y = dinv[row] * a;
            ((__bf16*)Cout)[(size_t)col * ldc + row] = (__bf16)y;
        }
    }
}

// ---------------------------------------------------------------------------
// 4-wide bf16 WMMA GEMM: each wave computes a 16x64 strip (4 col tiles) and
// reuses one A fragment per K-step -> loads/WMMA drops 4 -> 2.5 and the big
// (A) matrix is re-read far fewer times. Requires M%128==0, N%64==0, K%32==0.
// EPI 1: H[row*ldc+col] = bf16(sigmoid(elu(dinv[row]*acc))) (bf16 out)
// EPI 3: C[row*ldc+col] = elu(dinv[row]*acc)                (float out)
// ---------------------------------------------------------------------------
template <int EPI>
__global__ void __launch_bounds__(256)
k_gemm_wmma_x4(const __bf16* __restrict__ A, const __bf16* __restrict__ Bt,
               int K, const float* __restrict__ dinv, void* __restrict__ Cout,
               int ldc)
{
    int tid  = threadIdx.x;
    int wave = tid >> 5;
    int lane = tid & 31;
    int hlf  = (lane >> 4) & 1;
    int r16  = lane & 15;

    int tileRow = (blockIdx.y * 8 + wave) * 16;
    int colBase = blockIdx.x * 64;

    v8f acc0 = {}, acc1 = {}, acc2 = {}, acc3 = {};
    const __bf16* pa0 = A  + (size_t)(tileRow + r16) * K + hlf * 16;
    const __bf16* pb0 = Bt + (size_t)(colBase + r16) * K + hlf * 16;
    const size_t tstride = (size_t)16 * K;   // col-tile stride in Bt

#pragma unroll 2
    for (int k0 = 0; k0 < K; k0 += 32) {
        UA ua;
        ua.h[0] = *(const v8bf*)(pa0 + k0);
        ua.h[1] = *(const v8bf*)(pa0 + k0 + 8);
        UA b0, b1, b2, b3;
        b0.h[0] = *(const v8bf*)(pb0 + k0);
        b0.h[1] = *(const v8bf*)(pb0 + k0 + 8);
        b1.h[0] = *(const v8bf*)(pb0 + tstride + k0);
        b1.h[1] = *(const v8bf*)(pb0 + tstride + k0 + 8);
        b2.h[0] = *(const v8bf*)(pb0 + 2 * tstride + k0);
        b2.h[1] = *(const v8bf*)(pb0 + 2 * tstride + k0 + 8);
        b3.h[0] = *(const v8bf*)(pb0 + 3 * tstride + k0);
        b3.h[1] = *(const v8bf*)(pb0 + 3 * tstride + k0 + 8);
        acc0 = __builtin_amdgcn_wmma_f32_16x16x32_bf16(
            false, ua.v, false, b0.v, (short)0, acc0, false, false);
        acc1 = __builtin_amdgcn_wmma_f32_16x16x32_bf16(
            false, ua.v, false, b1.v, (short)0, acc1, false, false);
        acc2 = __builtin_amdgcn_wmma_f32_16x16x32_bf16(
            false, ua.v, false, b2.v, (short)0, acc2, false, false);
        acc3 = __builtin_amdgcn_wmma_f32_16x16x32_bf16(
            false, ua.v, false, b3.v, (short)0, acc3, false, false);
    }

    v8f accs[4] = {acc0, acc1, acc2, acc3};
#pragma unroll
    for (int t = 0; t < 4; ++t) {
        int col = colBase + t * 16 + r16;
#pragma unroll
        for (int v = 0; v < 8; ++v) {
            int row = tileRow + v + hlf * 8;
            float a = accs[t][v];
            float y = dinv[row] * a;
            if (EPI == 1) {
                float e = eluf(y);
                float h = 1.f / (1.f + __expf(-e));
                ((__bf16*)Cout)[(size_t)row * ldc + col] = (__bf16)h;
            } else { // EPI == 3
                ((float*)Cout)[(size_t)row * ldc + col] = eluf(y);
            }
        }
    }
}

// acc[i,c] (+)= (h2[i,c]-mean[c])*rstd[c]*g[c]+b[c];  one block per row
__global__ void __launch_bounds__(64)
k_bn_accum(const float* __restrict__ h2, const float* __restrict__ mean,
           const float* __restrict__ rstd, const float* __restrict__ g,
           const float* __restrict__ b, float* __restrict__ acc, int first)
{
    int i = blockIdx.x;
    int c = threadIdx.x;                 // OUTD == 64
    float v = (h2[(size_t)i * OUTD + c] - mean[c]) * rstd[c] * g[c] + b[c];
    size_t o = (size_t)i * OUTD + c;
    acc[o] = first ? v : (acc[o] + v);
}

__global__ void __launch_bounds__(256)
k_final_relu_mean(const float* __restrict__ acc, float* __restrict__ out, int total)
{
    int i = blockIdx.x * 256 + threadIdx.x;
    if (i < total) out[i] = fmaxf(acc[i] * (1.f / 3.f), 0.f);
}

// ---------------------------------------------------------------------------
// Workspace layout (bytes, 256-aligned)
// ---------------------------------------------------------------------------
static constexpr size_t OFF_MEAN1 = 0;                          // 256 f32
static constexpr size_t OFF_RSTD1 = 4096;                       // 256 f32
static constexpr size_t OFF_MEAN2 = 8192;                       // 64 f32
static constexpr size_t OFF_RSTD2 = 12288;                      // 64 f32
static constexpr size_t OFF_DINV  = 16384;                      // 4096 f32
static constexpr size_t OFF_XW1   = 32768;                      // 4096x128 f32  (2 MB)
static constexpr size_t OFF_H2    = OFF_XW1  + (size_t)BB * HID * 4;   // 4096x64 f32 (1 MB)
static constexpr size_t OFF_ACC   = OFF_H2   + (size_t)BB * OUTD * 4;  // 4096x64 f32 (1 MB)
static constexpr size_t OFF_XN    = OFF_ACC  + (size_t)BB * OUTD * 4;  // 4096x256 bf16 (2 MB)
static constexpr size_t OFF_W1T   = OFF_XN   + (size_t)BB * IN_DIM * 2;
static constexpr size_t OFF_W2T   = OFF_W1T  + (size_t)HID * IN_DIM * 2;
static constexpr size_t OFF_ZST   = OFF_W2T  + (size_t)OUTD * HID * 2; // 128x4096 bf16 (1 MB)
static constexpr size_t OFF_H1    = OFF_ZST  + (size_t)HID * BB * 2;   // 4096x128 bf16 (1 MB)
static constexpr size_t OFF_Z2ST  = OFF_H1   + (size_t)BB * HID * 2;   // 64x4096 bf16 (0.5 MB)
static constexpr size_t OFF_BA    = OFF_Z2ST + (size_t)OUTD * BB * 2;  // 4096x4096 bf16 (32 MB)
static constexpr size_t WS_NEEDED = OFF_BA + (size_t)BB * BB * 2;      // ~40.6 MB

extern "C" void kernel_launch(void* const* d_in, const int* in_sizes, int n_in,
                              void* d_out, int out_size, void* d_ws, size_t ws_size,
                              hipStream_t stream)
{
    const float* features = (const float*)d_in[0];
    const float* adj      = (const float*)d_in[1];
    const int*   bn       = (const int*)  d_in[2];
    const float* w1       = (const float*)d_in[3];
    const float* w2       = (const float*)d_in[4];
    const float* g1       = (const float*)d_in[5];
    const float* b1       = (const float*)d_in[6];
    const float* g2       = (const float*)d_in[7];
    const float* b2       = (const float*)d_in[8];
    float* out = (float*)d_out;
    char*  ws  = (char*)d_ws;
    if (ws_size < WS_NEEDED) return;

    float*  mean1 = (float*)(ws + OFF_MEAN1);
    float*  rstd1 = (float*)(ws + OFF_RSTD1);
    float*  mean2 = (float*)(ws + OFF_MEAN2);
    float*  rstd2 = (float*)(ws + OFF_RSTD2);
    float*  dinv  = (float*)(ws + OFF_DINV);
    float*  Xw1   = (float*)(ws + OFF_XW1);
    float*  h2    = (float*)(ws + OFF_H2);
    float*  acc   = (float*)(ws + OFF_ACC);
    __bf16* xn    = (__bf16*)(ws + OFF_XN);
    __bf16* w1t   = (__bf16*)(ws + OFF_W1T);
    __bf16* w2t   = (__bf16*)(ws + OFF_W2T);
    __bf16* Zst   = (__bf16*)(ws + OFF_ZST);
    __bf16* h1    = (__bf16*)(ws + OFF_H1);
    __bf16* Z2st  = (__bf16*)(ws + OFF_Z2ST);
    __bf16* bA    = (__bf16*)(ws + OFF_BA);

    // ---- shared-across-relations prologue ----
    k_col_stats<<<IN_DIM, 256, 0, stream>>>(features, bn, BB, IN_DIM, mean1, rstd1);
    k_bn_gather_bf16<<<BB, 256, 0, stream>>>(features, bn, mean1, rstd1, g1, b1, xn);
    k_transpose_bf16<<<HID, 256, 0, stream>>>(w1, w1t, IN_DIM, HID);
    k_transpose_bf16<<<OUTD, 256, 0, stream>>>(w2, w2t, HID, OUTD);
    // Xw1 = xn @ w1   (M=4096, N=128, K=256)
    k_gemm_wmma<0><<<dim3(HID / 16, BB / 128), 256, 0, stream>>>(
        xn, w1t, IN_DIM, nullptr, Xw1, HID);

    // ---- per relation ----
    for (int r = 0; r < RREL; ++r) {
        k_gather_adj<<<BB, 256, 0, stream>>>(adj, r, bn, bA, dinv);
        k_build_zst<<<BB, 128, 0, stream>>>(Xw1, dinv, Zst);
        // h1 = sigmoid(elu(dinv_row * (A @ Zst^T)))   (M=4096, N=128, K=4096)
        k_gemm_wmma_x4<1><<<dim3(HID / 64, BB / 128), 256, 0, stream>>>(
            bA, Zst, BB, dinv, h1, HID);
        // Z2st[n,j] = dinv[j] * (h1 @ w2)[j,n]        (M=4096, N=64, K=128)
        k_gemm_wmma<2><<<dim3(OUTD / 16, BB / 128), 256, 0, stream>>>(
            h1, w2t, HID, dinv, Z2st, BB);
        // h2 = elu(dinv_row * (A @ Z2st^T))           (M=4096, N=64, K=4096)
        k_gemm_wmma_x4<3><<<dim3(OUTD / 64, BB / 128), 256, 0, stream>>>(
            bA, Z2st, BB, dinv, h2, OUTD);
        k_col_stats<<<OUTD, 256, 0, stream>>>(h2, nullptr, BB, OUTD, mean2, rstd2);
        k_bn_accum<<<BB, 64, 0, stream>>>(h2, mean2, rstd2, g2, b2, acc, r == 0 ? 1 : 0);
    }

    int total = BB * OUTD;
    k_final_relu_mean<<<(total + 255) / 256, 256, 0, stream>>>(acc, out, total);
}
